// AttentionResblock_78228534329789
// MI455X (gfx1250) — compile-verified, matrix operand order checked by reading, not run on hardware
//
#include <hip/hip_runtime.h>
#include <hip/hip_bf16.h>

// ---------------------------------------------------------------------------
// Types for CDNA5 WMMA (wave32): v_wmma_f32_16x16x32_bf16
// ---------------------------------------------------------------------------
typedef __attribute__((ext_vector_type(16))) __bf16 bf16x16;
typedef __attribute__((ext_vector_type(8)))  float  f32x8;
typedef unsigned int   u32;
typedef unsigned short u16;
typedef __attribute__((ext_vector_type(4))) u32 u32x4;

union Frag {
    bf16x16 bf;
    u32x4   q[2];
};

__device__ __forceinline__ f32x8 zero8() {
    f32x8 z;
#pragma unroll
    for (int i = 0; i < 8; ++i) z[i] = 0.0f;
    return z;
}

__device__ __forceinline__ u16 f2bf(float x) {
    u32 u = __float_as_uint(x);
    u32 r = u + 0x7FFFu + ((u >> 16) & 1u);   // round-to-nearest-even
    return (u16)(r >> 16);
}

__device__ __forceinline__ f32x8 wmma_bf16(const Frag& a, const Frag& b, f32x8 c) {
    return __builtin_amdgcn_wmma_f32_16x16x32_bf16(false, a.bf, false, b.bf,
                                                   (short)0, c, false, false);
}

// CDNA5 async DMA: global -> LDS, tracked by ASYNCcnt (per-wave, in-order).
__device__ __forceinline__ void async_b128(const void* gptr, u32 lds_off) {
    asm volatile("global_load_async_to_lds_b128 %0, %1, off"
                 :: "v"(lds_off), "v"(gptr) : "memory");
}
#define WAIT_ASYNC(n) asm volatile("s_wait_asynccnt " #n ::: "memory")

// flat shared address: low 32 bits == LDS byte offset (aperture layout)
__device__ __forceinline__ u32 lds_off(const void* p) { return (u32)(size_t)p; }

#define N_STATE 1024
#define N_HEADS 16
#define D_HEAD  64
#define N_CTX   4080
#define T_PAD   4096

// ---------------------------------------------------------------------------
// 1. fp32 W[k][n] (1024x1024) -> bf16 Wt[n][k]
// ---------------------------------------------------------------------------
__global__ void k_transpose_bf16(const float* __restrict__ W, u16* __restrict__ Wt) {
    __shared__ u16 tile[32][33];
    const int bx = blockIdx.x * 32, by = blockIdx.y * 32;
    const int tx = threadIdx.x, ty = threadIdx.y;     // 32 x 8
#pragma unroll
    for (int i = 0; i < 4; ++i) {
        int k = by + ty + i * 8;
        tile[ty + i * 8][tx] = f2bf(W[(size_t)k * N_STATE + bx + tx]);
    }
    __syncthreads();
#pragma unroll
    for (int i = 0; i < 4; ++i) {
        int n = bx + ty + i * 8;
        Wt[(size_t)n * N_STATE + by + tx] = tile[tx][ty + i * 8];
    }
}

// ---------------------------------------------------------------------------
// 2. LayerNorm -> bf16, rows >= N_CTX zeroed (reference zero-pads post-matmul)
// ---------------------------------------------------------------------------
__global__ void k_layernorm(const float* __restrict__ m, const float* __restrict__ gamma,
                            const float* __restrict__ beta, u16* __restrict__ r) {
    const int t   = blockIdx.x;
    const int tid = threadIdx.x;                       // 256 threads
    u16* row_out = r + (size_t)t * N_STATE;
    if (t >= N_CTX) {
        for (int i = tid; i < N_STATE; i += 256) row_out[i] = 0;
        return;
    }
    const float* row = m + (size_t)t * N_STATE;
    float x[4], s = 0.0f, ss = 0.0f;
#pragma unroll
    for (int i = 0; i < 4; ++i) {
        x[i] = row[tid + i * 256];
        s  += x[i];
        ss += x[i] * x[i];
    }
#pragma unroll
    for (int off = 16; off > 0; off >>= 1) {
        s  += __shfl_xor(s,  off, 32);
        ss += __shfl_xor(ss, off, 32);
    }
    __shared__ float rs[8], rss[8];
    const int wave = tid >> 5, lane = tid & 31;
    if (lane == 0) { rs[wave] = s; rss[wave] = ss; }
    __syncthreads();
    s = 0.0f; ss = 0.0f;
#pragma unroll
    for (int w = 0; w < 8; ++w) { s += rs[w]; ss += rss[w]; }
    const float mu   = s * (1.0f / 1024.0f);
    const float var  = ss * (1.0f / 1024.0f) - mu * mu;
    const float rstd = rsqrtf(var + 1e-5f);
#pragma unroll
    for (int i = 0; i < 4; ++i) {
        int c = tid + i * 256;
        row_out[c] = f2bf((x[i] - mu) * rstd * gamma[c] + beta[c]);
    }
}

// ---------------------------------------------------------------------------
// Shared GEMM core: block = 256 thr (8 waves x 16 rows), tile M=128 N=64.
// Pipeline stage K=64: B tile (64x64 bf16, pitch 72 u16 -> 36-bank row shift,
// conflict-free b128 reads) async-staged to LDS, double buffered.
// 2 barriers per 8 WMMAs/wave; accumulators stay in one tuple (unroll 1).
// ---------------------------------------------------------------------------
#define KB_PITCH 72

__device__ __forceinline__ void issue_b_tile64(const u16* __restrict__ Wt, int n0, int k0,
                                               u16 (*bsm)[KB_PITCH]) {
    const int tid = threadIdx.x;       // 256 threads, 2 async instr each
#pragma unroll
    for (int rep = 0; rep < 2; ++rep) {
        const int cid = tid + rep * 256;       // 512 chunks: 64 rows x 8
        const int row = cid >> 3;
        const int c   = (cid & 7) * 8;
        async_b128(Wt + (size_t)(n0 + row) * N_STATE + k0 + c, lds_off(&bsm[row][c]));
    }
}

__device__ __forceinline__ void gemm_core(const u16* __restrict__ arow,
                                          const u16* __restrict__ Wt, int n0,
                                          int half, int ln,
                                          u16 (*bsm0)[KB_PITCH], u16 (*bsm1)[KB_PITCH],
                                          f32x8 (&acc)[4]) {
    issue_b_tile64(Wt, n0, 0, bsm0);
#pragma unroll 1
    for (int i = 0; i < N_STATE / 64; ++i) {
        const int k0 = i * 64;
        u16 (*cur)[KB_PITCH] = (i & 1) ? bsm1 : bsm0;
        if (i + 1 < N_STATE / 64) {
            issue_b_tile64(Wt, n0, k0 + 64, (i & 1) ? bsm0 : bsm1);
            WAIT_ASYNC(0x2);
        } else {
            WAIT_ASYNC(0x0);
        }
        __syncthreads();                       // stage i visible to all waves
#pragma unroll
        for (int s = 0; s < 2; ++s) {
            const int kk = k0 + s * 32;
            Frag a;
            a.q[0] = *(const u32x4*)(arow + kk + half * 8);
            a.q[1] = *(const u32x4*)(arow + kk + 16 + half * 8);
            Frag b[4];                         // hoist all B frags, then WMMA burst
#pragma unroll
            for (int nt = 0; nt < 4; ++nt) {
                const u16* bp = &cur[nt * 16 + ln][s * 32];
                b[nt].q[0] = *(const u32x4*)(bp + half * 8);
                b[nt].q[1] = *(const u32x4*)(bp + 16 + half * 8);
            }
#pragma unroll
            for (int nt = 0; nt < 4; ++nt) acc[nt] = wmma_bf16(a, b[nt], acc[nt]);
        }
        __syncthreads();                       // all reads done -> buffer reusable
    }
}

// ---------------------------------------------------------------------------
// 3. QKV projection GEMM (grid.x: 48 = 3 matrices x 16 n-tiles, grid.y: 32)
// ---------------------------------------------------------------------------
__global__ void k_qkv_gemm(const u16* __restrict__ A,
                           const u16* __restrict__ WqT, const u16* __restrict__ WkT,
                           const u16* __restrict__ WvT,
                           const float* __restrict__ bq, const float* __restrict__ bv,
                           u16* __restrict__ q_ws, u16* __restrict__ k_ws,
                           u16* __restrict__ vT_ws) {
    __shared__ __align__(16) u16 bsm[2][64][KB_PITCH];
    const int lane = threadIdx.x & 31;
    const int wave = threadIdx.x >> 5;
    const int half = lane >> 4;
    const int ln   = lane & 15;
    const int which = blockIdx.x >> 4;             // 0=q 1=k 2=v
    const int n0 = (blockIdx.x & 15) * 64;
    const int m0 = blockIdx.y * 128 + wave * 16;
    const u16* Wt = (which == 0) ? WqT : (which == 1) ? WkT : WvT;

    f32x8 acc[4];
#pragma unroll
    for (int nt = 0; nt < 4; ++nt) acc[nt] = zero8();

    gemm_core(A + (size_t)(m0 + ln) * N_STATE, Wt, n0, half, ln, bsm[0], bsm[1], acc);

#pragma unroll
    for (int nt = 0; nt < 4; ++nt) {
#pragma unroll
        for (int j = 0; j < 8; ++j) {
            const int t = m0 + j + 8 * half;               // C layout: M = j+8*half
            const int n = n0 + nt * 16 + ln;               //           N = lane&15
            const float v = acc[nt][j];
            if (which == 0) {
                q_ws[((size_t)(n >> 6) * T_PAD + t) * D_HEAD + (n & 63)] = f2bf(v + bq[n]);
            } else if (which == 1) {
                k_ws[((size_t)(n >> 6) * T_PAD + t) * D_HEAD + (n & 63)] = f2bf(v);
            } else {
                const float vv = (t < N_CTX) ? (v + bv[n]) : 0.0f;  // ref pads after +bv
                vT_ws[(size_t)n * T_PAD + t] = f2bf(vv);
            }
        }
    }
}

// ---------------------------------------------------------------------------
// 4. Flash attention: block = (head, 64 q-rows), 4 waves x 16 q-rows.
//    K tile (32x64, pitch 72) and V tile (64x32, pitch 40) async-staged to
//    LDS, double buffered, shared by the 4 waves.
// ---------------------------------------------------------------------------
#define V_PITCH 40

__device__ __forceinline__ void issue_kv_tile(const u16* __restrict__ kh,
                                              const u16* __restrict__ vh,
                                              int kt,
                                              u16 (*kb)[KB_PITCH], u16 (*vb)[V_PITCH]) {
    const int tid = threadIdx.x;       // 128 threads, 4 async instr each
#pragma unroll
    for (int rep = 0; rep < 2; ++rep) {
        const int cid = tid + rep * 128;           // 256 chunks each tile
        const int kr = cid >> 3, kc = (cid & 7) * 8;     // K: 32 rows x 8
        async_b128(kh + (size_t)(kt + kr) * D_HEAD + kc, lds_off(&kb[kr][kc]));
        const int vr = cid >> 2, vc = (cid & 3) * 8;     // V: 64 rows x 4
        async_b128(vh + (size_t)vr * T_PAD + kt + vc, lds_off(&vb[vr][vc]));
    }
}

__global__ void k_attention(const u16* __restrict__ q_ws, const u16* __restrict__ k_ws,
                            const u16* __restrict__ vT_ws, const float* __restrict__ bias,
                            u16* __restrict__ o_ws) {
    __shared__ __align__(16) u16 kbuf[2][32][KB_PITCH];
    __shared__ __align__(16) u16 vbuf[2][64][V_PITCH];
    __shared__ __align__(16) u16 pbuf[4][16][32];
    const int lane = threadIdx.x & 31;
    const int wave = threadIdx.x >> 5;
    const int half = lane >> 4;
    const int ln   = lane & 15;
    const int h  = blockIdx.x;
    const int q0 = blockIdx.y * 64 + wave * 16;

    const u16* kh = k_ws  + (size_t)h * T_PAD * D_HEAD;
    const u16* vh = vT_ws + (size_t)h * D_HEAD * T_PAD;

    // Q fragments (16x64 as two 16x32 A-fragments), resident for whole loop
    const u16* qbase = q_ws + ((size_t)h * T_PAD + q0 + ln) * D_HEAD;
    Frag aq0, aq1;
    aq0.q[0] = *(const u32x4*)(qbase + half * 8);
    aq0.q[1] = *(const u32x4*)(qbase + 16 + half * 8);
    aq1.q[0] = *(const u32x4*)(qbase + 32 + half * 8);
    aq1.q[1] = *(const u32x4*)(qbase + 48 + half * 8);

    f32x8 o[4];
#pragma unroll
    for (int dt = 0; dt < 4; ++dt) o[dt] = zero8();
    float mrun[8], lrun[8];
#pragma unroll
    for (int j = 0; j < 8; ++j) { mrun[j] = -3.0e38f; lrun[j] = 0.0f; }

    const float scale = 0.125f;   // 1/sqrt(64)

    issue_kv_tile(kh, vh, 0, kbuf[0], vbuf[0]);

    const int NSTEP = T_PAD / 32;
#pragma unroll 1
    for (int i = 0; i < NSTEP; ++i) {
        const int kt  = i * 32;
        const int buf = i & 1;
        if (i + 1 < NSTEP) {
            issue_kv_tile(kh, vh, kt + 32, kbuf[buf ^ 1], vbuf[buf ^ 1]);
            WAIT_ASYNC(0x4);
        } else {
            WAIT_ASYNC(0x0);
        }
        __syncthreads();                       // tile i ready for all 4 waves

        // ---- K fragments hoisted, then S = Q @ K^T as a WMMA burst
        Frag kf[4];
#pragma unroll
        for (int nt = 0; nt < 2; ++nt) {
            const u16* kp = &kbuf[buf][nt * 16 + ln][0];
            kf[2 * nt + 0].q[0] = *(const u32x4*)(kp + half * 8);
            kf[2 * nt + 0].q[1] = *(const u32x4*)(kp + 16 + half * 8);
            kf[2 * nt + 1].q[0] = *(const u32x4*)(kp + 32 + half * 8);
            kf[2 * nt + 1].q[1] = *(const u32x4*)(kp + 48 + half * 8);
        }
        f32x8 s[2];
#pragma unroll
        for (int nt = 0; nt < 2; ++nt) {
            f32x8 t = zero8();
            t = wmma_bf16(aq0, kf[2 * nt + 0], t);
            t = wmma_bf16(aq1, kf[2 * nt + 1], t);
            s[nt] = t;
        }
        // ---- scale + additive bias (global stream; resident in L2 across heads)
#pragma unroll
        for (int nt = 0; nt < 2; ++nt)
#pragma unroll
            for (int j = 0; j < 8; ++j) {
                const int row = q0 + j + 8 * half;
                const int col = kt + nt * 16 + ln;
                s[nt][j] = s[nt][j] * scale + bias[(size_t)row * T_PAD + col];
            }
        // ---- online softmax (rows live across the 16 lanes of a half)
        float alpha[8];
#pragma unroll
        for (int j = 0; j < 8; ++j) {
            float v = fmaxf(s[0][j], s[1][j]);
#pragma unroll
            for (int off = 8; off > 0; off >>= 1) v = fmaxf(v, __shfl_xor(v, off, 16));
            const float nm = fmaxf(mrun[j], v);
            const float a  = __expf(mrun[j] - nm);
            const float p0 = __expf(s[0][j] - nm);
            const float p1 = __expf(s[1][j] - nm);
            float rsum = p0 + p1;
#pragma unroll
            for (int off = 8; off > 0; off >>= 1) rsum += __shfl_xor(rsum, off, 16);
            lrun[j] = lrun[j] * a + rsum;
            mrun[j] = nm;
            alpha[j] = a;
            s[0][j] = p0;
            s[1][j] = p1;
        }
#pragma unroll
        for (int dt = 0; dt < 4; ++dt)
#pragma unroll
            for (int j = 0; j < 8; ++j) o[dt][j] *= alpha[j];

        // ---- V fragments into registers BEFORE releasing the buffer
        Frag vb[4];
#pragma unroll
        for (int dt = 0; dt < 4; ++dt) {
            const u16* vp = &vbuf[buf][dt * 16 + ln][0];
            vb[dt].q[0] = *(const u32x4*)(vp + half * 8);
            vb[dt].q[1] = *(const u32x4*)(vp + 16 + half * 8);
        }
        // ---- P: C-layout -> A-layout via per-wave LDS bounce
#pragma unroll
        for (int nt = 0; nt < 2; ++nt)
#pragma unroll
            for (int j = 0; j < 8; ++j)
                pbuf[wave][j + 8 * half][nt * 16 + ln] = f2bf(s[nt][j]);
        __syncthreads();        // pbuf ready; all K/V reads of buf complete
        Frag p;
        p.q[0] = *(const u32x4*)(&pbuf[wave][ln][half * 8]);
        p.q[1] = *(const u32x4*)(&pbuf[wave][ln][16 + half * 8]);

        // ---- O += P @ V
#pragma unroll
        for (int dt = 0; dt < 4; ++dt) o[dt] = wmma_bf16(p, vb[dt], o[dt]);
    }
    // ---- normalize, write [t][1024] bf16
#pragma unroll
    for (int dt = 0; dt < 4; ++dt)
#pragma unroll
        for (int j = 0; j < 8; ++j) {
            const int row = q0 + j + 8 * half;
            const int col = h * D_HEAD + dt * 16 + ln;
            o_ws[(size_t)row * N_STATE + col] = f2bf(o[dt][j] / lrun[j]);
        }
}

// ---------------------------------------------------------------------------
// 5. Output projection + residual: out = m + attn @ Wc + bc (fp32, t < 4080)
// ---------------------------------------------------------------------------
__global__ void k_out_gemm(const u16* __restrict__ A, const u16* __restrict__ WcT,
                           const float* __restrict__ bc, const float* __restrict__ m_in,
                           float* __restrict__ out) {
    __shared__ __align__(16) u16 bsm[2][64][KB_PITCH];
    const int lane = threadIdx.x & 31;
    const int wave = threadIdx.x >> 5;
    const int half = lane >> 4;
    const int ln   = lane & 15;
    const int n0 = blockIdx.x * 64;
    const int m0 = blockIdx.y * 128 + wave * 16;

    f32x8 acc[4];
#pragma unroll
    for (int nt = 0; nt < 4; ++nt) acc[nt] = zero8();

    gemm_core(A + (size_t)(m0 + ln) * N_STATE, WcT, n0, half, ln, bsm[0], bsm[1], acc);

#pragma unroll
    for (int nt = 0; nt < 4; ++nt)
#pragma unroll
        for (int j = 0; j < 8; ++j) {
            const int t = m0 + j + 8 * half;
            if (t < N_CTX) {
                const int n = n0 + nt * 16 + ln;
                out[(size_t)t * N_STATE + n] =
                    m_in[(size_t)t * N_STATE + n] + acc[nt][j] + bc[n];
            }
        }
}

// ---------------------------------------------------------------------------
// host-side orchestration
// ---------------------------------------------------------------------------
extern "C" void kernel_launch(void* const* d_in, const int* in_sizes, int n_in,
                              void* d_out, int out_size, void* d_ws, size_t ws_size,
                              hipStream_t stream) {
    const float* m     = (const float*)d_in[0];
    const float* bias  = (const float*)d_in[1];
    const float* gamma = (const float*)d_in[2];
    const float* beta  = (const float*)d_in[3];
    const float* Wq    = (const float*)d_in[4];
    const float* bq    = (const float*)d_in[5];
    const float* Wk    = (const float*)d_in[6];
    const float* Wv    = (const float*)d_in[7];
    const float* bv    = (const float*)d_in[8];
    const float* Wc    = (const float*)d_in[9];
    const float* bc    = (const float*)d_in[10];
    float* out = (float*)d_out;

    char* ws = (char*)d_ws;
    const size_t SZ_ACT = (size_t)T_PAD * N_STATE * sizeof(u16);   // 8 MB
    const size_t SZ_W   = (size_t)N_STATE * N_STATE * sizeof(u16); // 2 MB
    u16* r_bf  = (u16*)(ws);
    u16* wqT   = (u16*)(ws + SZ_ACT);
    u16* wkT   = (u16*)(ws + SZ_ACT + SZ_W);
    u16* wvT   = (u16*)(ws + SZ_ACT + 2 * SZ_W);
    u16* wcT   = (u16*)(ws + SZ_ACT + 3 * SZ_W);
    u16* q_ws  = (u16*)(ws + SZ_ACT + 4 * SZ_W);
    u16* k_ws  = (u16*)(ws + 2 * SZ_ACT + 4 * SZ_W);
    u16* vT_ws = (u16*)(ws + 3 * SZ_ACT + 4 * SZ_W);
    u16* o_ws  = (u16*)(ws + 4 * SZ_ACT + 4 * SZ_W);

    dim3 tb(32, 8);
    dim3 tg(32, 32);
    k_transpose_bf16<<<tg, tb, 0, stream>>>(Wq, wqT);
    k_transpose_bf16<<<tg, tb, 0, stream>>>(Wk, wkT);
    k_transpose_bf16<<<tg, tb, 0, stream>>>(Wv, wvT);
    k_transpose_bf16<<<tg, tb, 0, stream>>>(Wc, wcT);

    k_layernorm<<<T_PAD, 256, 0, stream>>>(m, gamma, beta, r_bf);

    k_qkv_gemm<<<dim3(48, 32), 256, 0, stream>>>(r_bf, wqT, wkT, wvT, bq, bv,
                                                 q_ws, k_ws, vT_ws);

    k_attention<<<dim3(N_HEADS, T_PAD / 64), 128, 0, stream>>>(q_ws, k_ws, vT_ws,
                                                               bias, o_ws);

    k_out_gemm<<<dim3(16, 32), 256, 0, stream>>>(o_ws, wcT, bc, m, out);
}
